// ManualCNN_82575041233118
// MI455X (gfx1250) — matrix operand order repaired and checked by reading, not compiled, not run on hardware
//
#include <hip/hip_runtime.h>
#include <hip/hip_bf16.h>

typedef __attribute__((ext_vector_type(16))) _Float16 v16h;
typedef __attribute__((ext_vector_type(8)))  float    v8f;
typedef __attribute__((ext_vector_type(4)))  unsigned u32x4;
typedef __attribute__((ext_vector_type(8)))  unsigned u32x8;

// A-fragment K index for v_wmma_f32_16x16x32_f16 (CDNA5 ISA 7.12.2):
// lanes 0-15: V0..V3 hold K 0..7, V4..V7 hold K 16..23; lanes 16-31: +8.
__device__ __forceinline__ int a_klocal(int e, int lane) {
  int v = e >> 1, p = e & 1, h = lane >> 4;
  return ((v >> 2) << 4) | (h << 3) | ((v & 3) << 1) | p;
}

// Issue a TDM DMA of `nbytes16` (multiple of 2, expressed in elements of 2B)
// from global `gaddr` into LDS byte offset `ldsb`. 2-group D# per ISA 08 §8:
// 1-row 2D tensor, data_size=2B, tile = whole row. Tracked by TENSORcnt.
__device__ __forceinline__ void tdm_load_row_f16(unsigned long long gaddr,
                                                 unsigned ldsb, unsigned nelem) {
  u32x4 g0;
  g0[0] = 1u;                                   // count=1, user descriptor
  g0[1] = ldsb;                                 // lds_addr (bytes)
  g0[2] = (unsigned)gaddr;                      // global_addr[31:0]
  g0[3] = ((unsigned)(gaddr >> 32) & 0x01FFFFFFu) | 0x80000000u; // [56:32]|type=2
  u32x8 g1;
  g1[0] = 1u << 16;                             // data_size=1 (2 bytes)
  g1[1] = (nelem & 0xFFFFu) << 16;              // tensor_dim0[15:0]
  g1[2] = (nelem >> 16) | (1u << 16);           // tensor_dim0[31:16], tensor_dim1=1
  g1[3] = (nelem & 0xFFFFu) << 16;              // tile_dim0 = nelem
  g1[4] = 1u;                                   // tile_dim1=1, tile_dim2=0
  g1[5] = nelem;                                // tensor_dim0_stride[31:0]
  g1[6] = 0u;
  g1[7] = 0u;
  asm volatile("tensor_load_to_lds %0, %1" :: "s"(g0), "s"(g1) : "memory");
}

// ---------------- weight prep: fp32 -> f16, K padded to mult of 32 ----------
__global__ void prep_weights(const float* __restrict__ w2,
                             const float* __restrict__ w3,
                             const float* __restrict__ fcw,
                             _Float16* __restrict__ w2h,   // [32][160]
                             _Float16* __restrict__ w3h,   // [64][320]
                             _Float16* __restrict__ fcwh)  // [16][1024]
{
  int i = blockIdx.x * blockDim.x + threadIdx.x;
  if (i < 32 * 160) {
    int m = i / 160, k = i % 160;
    w2h[i] = (_Float16)((k < 144) ? w2[m * 144 + k] : 0.f);
  }
  int j = i - 32 * 160;
  if (j >= 0 && j < 64 * 320) {
    int m = j / 320, k = j % 320;
    w3h[j] = (_Float16)((k < 288) ? w3[m * 288 + k] : 0.f);
  }
  int l = i - 32 * 160 - 64 * 320;
  if (l >= 0 && l < 16 * 1024) {
    int m = l / 1024, k = l % 1024;
    fcwh[l] = (_Float16)((m < 10) ? fcw[m * 1024 + k] : 0.f);
  }
}

// ---------------- conv1 (3->16, 32x32) + relu + 2x2 pool, direct VALU -------
__global__ void __launch_bounds__(256) conv1_pool(
    const float* __restrict__ x, const float* __restrict__ w1,
    const float* __restrict__ b1, _Float16* __restrict__ h1)
{
  int idx = blockIdx.x * blockDim.x + threadIdx.x;   // B*16*16*16
  if (idx >= 4096 * 16 * 16 * 16) return;
  int b  = idx >> 12;
  int c  = (idx >> 8) & 15;
  int py = (idx >> 4) & 15;
  int px = idx & 15;
  const float* xb = x + (size_t)b * 3 * 32 * 32;
  const float* wc = w1 + c * 27;
  float bias = b1[c];
  float best = 0.f;                                   // max of ReLU >= 0
  #pragma unroll
  for (int sy = 0; sy < 2; ++sy)
    #pragma unroll
    for (int sx = 0; sx < 2; ++sx) {
      int oy = 2 * py + sy, ox = 2 * px + sx;
      float acc = bias;
      #pragma unroll
      for (int ci = 0; ci < 3; ++ci)
        #pragma unroll
        for (int di = 0; di < 3; ++di) {
          int iy = oy + di - 1;
          if (iy < 0 || iy > 31) continue;
          #pragma unroll
          for (int dj = 0; dj < 3; ++dj) {
            int ix = ox + dj - 1;
            if (ix < 0 || ix > 31) continue;
            acc += wc[ci * 9 + di * 3 + dj] * xb[ci * 1024 + iy * 32 + ix];
          }
        }
      best = fmaxf(best, acc);
    }
  h1[idx] = (_Float16)best;                           // [B,16,16,16] f16 NCHW
}

// ---------------- conv2 (16->32, 16x16) implicit GEMM via WMMA --------------
// Activations staged by one TDM DMA (tensor_load_to_lds, TENSORcnt).
// M=32 (2 tiles), K=16*9=144 -> 160 (5 tiles), N=256 positions (16 tiles).
__global__ void __launch_bounds__(128) conv2_pool(
    const _Float16* __restrict__ h1, const _Float16* __restrict__ w2h,
    const float* __restrict__ b2, _Float16* __restrict__ h2)
{
  __shared__ _Float16 sIn[16 * 16 * 16];   // 8 KB
  __shared__ _Float16 sOut[32 * 16 * 16];  // 16 KB
  int b = blockIdx.x;
  int tid = threadIdx.x, lane = tid & 31, wave = tid >> 5;
  const _Float16* src = h1 + (size_t)b * 4096;
  __builtin_prefetch(h1 + (size_t)(b + 1) * 4096, 0, 1);

  if (tid < 32) {                           // one TDM issue per workgroup
    tdm_load_row_f16((unsigned long long)src,
                     (unsigned)(unsigned long long)(void*)sIn, 4096u);
    __builtin_amdgcn_s_wait_tensorcnt(0);
  }
  __syncthreads();

  // Preload all A fragments (weights), constant over N tiles.
  v16h A[2][5];
  #pragma unroll
  for (int mt = 0; mt < 2; ++mt)
    #pragma unroll
    for (int kt = 0; kt < 5; ++kt) {
      int m = mt * 16 + (lane & 15);
      v16h a;
      #pragma unroll
      for (int e = 0; e < 16; ++e)
        a[e] = w2h[m * 160 + kt * 32 + a_klocal(e, lane)];
      A[mt][kt] = a;
    }

  for (int nt = wave; nt < 16; nt += 4) {              // one output row per tile
    v8f acc0 = {}; v8f acc1 = {};
    #pragma unroll
    for (int kt = 0; kt < 5; ++kt) {
      int n = nt * 16 + (lane & 15);                   // B layout: column n = lane % 16
      v16h bf;
      #pragma unroll
      for (int e = 0; e < 16; ++e) {
        int k = kt * 32 + a_klocal(e, lane);
        bf[e] = (_Float16)0.f;
        if (k < 144) {
          int ci = k / 9, r = k % 9, di = r / 3, dj = r % 3;
          int iy = (n >> 4) + di - 1;
          int ix = (n & 15) + dj - 1;
          if (iy >= 0 && iy < 16 && ix >= 0 && ix < 16)
            bf[e] = sIn[ci * 256 + iy * 16 + ix];
        }
      }
      acc0 = __builtin_amdgcn_wmma_f32_16x16x32_f16(false, A[0][kt], false, bf,
                                                    (short)0, acc0, false, false);
      acc1 = __builtin_amdgcn_wmma_f32_16x16x32_f16(false, A[1][kt], false, bf,
                                                    (short)0, acc1, false, false);
    }
    int n = nt * 16 + (lane & 15);
    int hh = lane >> 4;
    #pragma unroll
    for (int ve = 0; ve < 8; ++ve) {
      int m0 = ve + 8 * hh;                            // C/D layout
      sOut[m0 * 256 + n]        = (_Float16)fmaxf(acc0[ve] + b2[m0], 0.f);
      sOut[(16 + m0) * 256 + n] = (_Float16)fmaxf(acc1[ve] + b2[16 + m0], 0.f);
    }
  }
  __syncthreads();
  __builtin_amdgcn_s_cluster_barrier();                // NOP outside a cluster

  _Float16* dst = h2 + (size_t)b * 2048;               // [B,32,8,8]
  for (int i = tid; i < 2048; i += 128) {
    int c = i >> 6, rem = i & 63, py = rem >> 3, px = rem & 7;
    const _Float16* p = &sOut[c * 256 + (py * 2) * 16 + px * 2];
    float m = fmaxf(fmaxf((float)p[0], (float)p[1]),
                    fmaxf((float)p[16], (float)p[17]));
    dst[i] = (_Float16)m;
  }
}

// ---------------- conv3 (32->64, 8x8) implicit GEMM via WMMA ----------------
// Activations staged by async global->LDS DMA (ASYNCcnt, no VGPR round trip).
// M=64 (4 tiles), K=32*9=288 -> 320 (10 tiles), N=64 positions (4 tiles).
__global__ void __launch_bounds__(128) conv3_pool(
    const _Float16* __restrict__ h2, const _Float16* __restrict__ w3h,
    const float* __restrict__ b3, _Float16* __restrict__ h3)
{
  __shared__ _Float16 sIn[32 * 64];    // 4 KB
  __shared__ _Float16 sOut[64 * 64];   // 8 KB
  int b = blockIdx.x;
  int tid = threadIdx.x, lane = tid & 31, wave = tid >> 5;
  const _Float16* src = h2 + (size_t)b * 2048;
  __builtin_prefetch(h2 + (size_t)(b + 1) * 2048, 0, 1);

  unsigned sInBase = (unsigned)(unsigned long long)(void*)sIn;
  #pragma unroll
  for (int i = 0; i < 2; ++i) {                     // 256 x 16B = 4 KB
    int t = tid + i * 128;
    unsigned ldsb = sInBase + t * 16;
    unsigned long long ga = (unsigned long long)src + t * 16;
    asm volatile("global_load_async_to_lds_b128 %0, %1, off"
                 :: "v"(ldsb), "v"(ga) : "memory");
  }
  asm volatile("s_wait_asynccnt 0" ::: "memory");
  __syncthreads();

  int nt = wave;                                       // 4 waves, 4 N tiles
  v8f acc[4] = {};
  for (int kt = 0; kt < 10; ++kt) {
    int nb = nt * 16 + (lane & 15);
    v16h bf;
    #pragma unroll
    for (int e = 0; e < 16; ++e) {
      int k = kt * 32 + a_klocal(e, lane);
      bf[e] = (_Float16)0.f;
      if (k < 288) {
        int ci = k / 9, r = k % 9, di = r / 3, dj = r % 3;
        int oy = nb >> 3, ox = nb & 7;
        int iy = oy + di - 1, ix = ox + dj - 1;
        if (iy >= 0 && iy < 8 && ix >= 0 && ix < 8)
          bf[e] = sIn[ci * 64 + iy * 8 + ix];
      }
    }
    #pragma unroll
    for (int mt = 0; mt < 4; ++mt) {
      int m = mt * 16 + (lane & 15);
      v16h a;
      #pragma unroll
      for (int e = 0; e < 16; ++e)
        a[e] = w3h[m * 320 + kt * 32 + a_klocal(e, lane)];
      acc[mt] = __builtin_amdgcn_wmma_f32_16x16x32_f16(false, a, false, bf,
                                                       (short)0, acc[mt], false, false);
    }
  }
  int n = nt * 16 + (lane & 15);
  int hh = lane >> 4;
  #pragma unroll
  for (int mt = 0; mt < 4; ++mt)
    #pragma unroll
    for (int ve = 0; ve < 8; ++ve) {
      int m = mt * 16 + ve + 8 * hh;
      sOut[m * 64 + n] = (_Float16)fmaxf(acc[mt][ve] + b3[m], 0.f);
    }
  __syncthreads();

  _Float16* dst = h3 + (size_t)b * 1024;               // [B,1024] flatten (c,y,x)
  for (int i = tid; i < 1024; i += 128) {
    int c = i >> 4, rem = i & 15, py = rem >> 2, px = rem & 3;
    const _Float16* p = &sOut[c * 64 + (py * 2) * 8 + px * 2];
    float m = fmaxf(fmaxf((float)p[0], (float)p[1]),
                    fmaxf((float)p[8], (float)p[9]));
    dst[i] = (_Float16)m;
  }
}

// ---------------- FC: [4096,1024] x [1024,10]^T + b, WMMA GEMM --------------
__global__ void __launch_bounds__(256) fc_kernel(
    const _Float16* __restrict__ h3, const _Float16* __restrict__ fcwh,
    const float* __restrict__ fcb, float* __restrict__ out)
{
  int tid = threadIdx.x, lane = tid & 31, wave = tid >> 5;
  int bt = blockIdx.x * 8 + wave;                      // 256 batch tiles of 16
  int nbase = bt * 16;
  v8f acc = {};
  for (int kt = 0; kt < 32; ++kt) {
    int m = lane & 15;
    v16h a;
    #pragma unroll
    for (int e = 0; e < 16; ++e)
      a[e] = fcwh[m * 1024 + kt * 32 + a_klocal(e, lane)];
    v16h bf;
    #pragma unroll
    for (int e = 0; e < 16; ++e)
      bf[e] = h3[(size_t)(nbase + (lane & 15)) * 1024 + kt * 32 + a_klocal(e, lane)];
    acc = __builtin_amdgcn_wmma_f32_16x16x32_f16(false, a, false, bf,
                                                 (short)0, acc, false, false);
  }
  int n = nbase + (lane & 15);
  int hh = lane >> 4;
  #pragma unroll
  for (int ve = 0; ve < 8; ++ve) {
    int m = ve + 8 * hh;
    if (m < 10) out[n * 10 + m] = acc[ve] + fcb[m];
  }
}

// ---------------------------------------------------------------------------
extern "C" void kernel_launch(void* const* d_in, const int* in_sizes, int n_in,
                              void* d_out, int out_size, void* d_ws, size_t ws_size,
                              hipStream_t stream) {
  const float* x    = (const float*)d_in[0];
  const float* w1   = (const float*)d_in[1];
  const float* b1   = (const float*)d_in[2];
  const float* w2   = (const float*)d_in[3];
  const float* b2   = (const float*)d_in[4];
  const float* w3   = (const float*)d_in[5];
  const float* b3   = (const float*)d_in[6];
  const float* fcw  = (const float*)d_in[7];
  const float* fcb  = (const float*)d_in[8];
  float* out = (float*)d_out;

  char* ws = (char*)d_ws;
  _Float16* w2h  = (_Float16*)(ws);                    // 32*160*2   = 10240
  _Float16* w3h  = (_Float16*)(ws + 10240);            // 64*320*2   = 40960
  _Float16* fcwh = (_Float16*)(ws + 51200);            // 16*1024*2  = 32768
  _Float16* h1   = (_Float16*)(ws + 83968);            // 4096*4096*2 = 33554432
  _Float16* h2   = (_Float16*)(ws + 83968 + 33554432);         // 16777216
  _Float16* h3   = (_Float16*)(ws + 83968 + 33554432 + 16777216); // 8388608

  prep_weights<<<(32 * 160 + 64 * 320 + 16 * 1024 + 255) / 256, 256, 0, stream>>>(
      w2, w3, fcw, w2h, w3h, fcwh);
  conv1_pool<<<(4096 * 4096) / 256, 256, 0, stream>>>(x, w1, b1, h1);
  conv2_pool<<<4096, 128, 0, stream>>>(h1, w2h, b2, h2);
  conv3_pool<<<4096, 128, 0, stream>>>(h2, w3h, b3, h3);
  fc_kernel<<<32, 256, 0, stream>>>(h3, fcwh, fcb, out);
}
